// AttentionHead_39926015984058
// MI455X (gfx1250) — matrix operand order repaired and checked by reading, not compiled, not run on hardware
//
#include <hip/hip_runtime.h>
#include <hip/hip_bf16.h>
#include <stdint.h>

#define D_EMB 512
#define HEAD  64
#define NB    8
#define NSEQ  2048
#define ROWS  (NB * NSEQ)   // 16384

typedef __attribute__((ext_vector_type(16))) __bf16 bf16x16;
typedef __attribute__((ext_vector_type(8)))  __bf16 bf16x8;
typedef __attribute__((ext_vector_type(8)))  float  v8f;

static __device__ inline v8f wmma_bf16(bf16x16 a, bf16x16 b, v8f c) {
  // 8 args: (neg_a, A, neg_b, B, c_mod, C, reuse_a, reuse_b)
  return __builtin_amdgcn_wmma_f32_16x16x32_bf16(false, a, false, b, (short)0, c,
                                                 false, false);
}

static __device__ inline bf16x16 cat8(bf16x8 lo, bf16x8 hi) {
  return __builtin_shufflevector(lo, hi, 0, 1, 2, 3, 4, 5, 6, 7,
                                 8, 9, 10, 11, 12, 13, 14, 15);
}

// ---------------------------------------------------------------------------
// Kernel 0: one-off transpose W f32[512][64] -> Wt bf16[64][512] so that
// projection B-fragments become single contiguous 32B loads.
// ---------------------------------------------------------------------------
__global__ __launch_bounds__(256)
void wtrans_kernel(const float* __restrict__ W, __bf16* __restrict__ Wt)
{
  const int idx = blockIdx.x * 256 + threadIdx.x;   // 0 .. 512*64-1
  const int k = idx >> 6;          // 0..511
  const int n = idx & 63;          // 0..63
  Wt[(size_t)n * D_EMB + k] = (__bf16)W[(size_t)k * HEAD + n];
}

// ---------------------------------------------------------------------------
// Kernel 1: P = X @ W_key  (X is one of query/key/value input, f32 [ROWS,512]).
// which==0 -> Q (pre-scaled 1/sqrt(512)), which==1 -> K (row major),
// which==2 -> V stored transposed [64][ROWS].
// One wave handles a 16-row x 64-col tile; K-dim consumed 32 at a time.
// ---------------------------------------------------------------------------
__global__ __launch_bounds__(32)
void proj_kernel(const float* __restrict__ Xq, const float* __restrict__ Xk,
                 const float* __restrict__ Xv, const __bf16* __restrict__ Wt,
                 __bf16* __restrict__ Qo, __bf16* __restrict__ Ko,
                 __bf16* __restrict__ Vt)
{
  const int lane = threadIdx.x;
  const int n16  = lane & 15;   // A-row / B-col / CD-col index
  const int kh   = lane >> 4;   // lane half
  const int which = blockIdx.y;
  const int row0  = blockIdx.x * 16;

  const float* X = (which == 0) ? Xq : (which == 1) ? Xk : Xv;
  const float* xrow = X + (size_t)(row0 + n16) * D_EMB;

  // B-fragment base pointers: lane-half kh holds K = k0 + kh*16 + {0..15},
  // contiguous in the transposed bf16 weights.
  const __bf16* wcol[4];
  #pragma unroll
  for (int nt = 0; nt < 4; ++nt)
    wcol[nt] = Wt + (size_t)(nt * 16 + n16) * D_EMB + kh * 16;

  v8f acc[4] = { {}, {}, {}, {} };

  for (int k0 = 0; k0 < D_EMB; k0 += 32) {
    // A fragment: elements [0..15] = K k0+kh*16+{0..15}
    const float* xp = xrow + k0 + kh * 16;
    float4 c0 = *(const float4*)(xp);
    float4 c1 = *(const float4*)(xp + 4);
    float4 c2 = *(const float4*)(xp + 8);
    float4 c3 = *(const float4*)(xp + 12);
    bf16x16 a;
    a[0]  = (__bf16)c0.x; a[1]  = (__bf16)c0.y; a[2]  = (__bf16)c0.z; a[3]  = (__bf16)c0.w;
    a[4]  = (__bf16)c1.x; a[5]  = (__bf16)c1.y; a[6]  = (__bf16)c1.z; a[7]  = (__bf16)c1.w;
    a[8]  = (__bf16)c2.x; a[9]  = (__bf16)c2.y; a[10] = (__bf16)c2.z; a[11] = (__bf16)c2.w;
    a[12] = (__bf16)c3.x; a[13] = (__bf16)c3.y; a[14] = (__bf16)c3.z; a[15] = (__bf16)c3.w;

    #pragma unroll
    for (int nt = 0; nt < 4; ++nt) {
      bf16x16 bb = *(const bf16x16*)(wcol[nt] + k0);
      acc[nt] = wmma_bf16(a, bb, acc[nt]);
    }
  }

  const float qscale = 0.044194173824159216f; // 1/sqrt(512)
  if (which == 0) {
    #pragma unroll
    for (int nt = 0; nt < 4; ++nt)
      #pragma unroll
      for (int r = 0; r < 8; ++r)
        Qo[(size_t)(row0 + r + 8 * kh) * HEAD + nt * 16 + n16] =
            (__bf16)(acc[nt][r] * qscale);
  } else if (which == 1) {
    #pragma unroll
    for (int nt = 0; nt < 4; ++nt)
      #pragma unroll
      for (int r = 0; r < 8; ++r)
        Ko[(size_t)(row0 + r + 8 * kh) * HEAD + nt * 16 + n16] = (__bf16)acc[nt][r];
  } else {
    #pragma unroll
    for (int nt = 0; nt < 4; ++nt)
      #pragma unroll
      for (int r = 0; r < 8; ++r)
        Vt[(size_t)(nt * 16 + n16) * ROWS + row0 + r + 8 * kh] = (__bf16)acc[nt][r];
  }
}

// ---------------------------------------------------------------------------
// Kernel 2: flash attention.  One wave = one (batch, 16-query tile).
// Keys processed in blocks of 32; online softmax; O accumulated in f32.
// ---------------------------------------------------------------------------
__global__ __launch_bounds__(32)
void attn_kernel(const __bf16* __restrict__ Q, const __bf16* __restrict__ K,
                 const __bf16* __restrict__ Vt, const unsigned char* __restrict__ pad,
                 const int* __restrict__ maskflag, float* __restrict__ out)
{
  __shared__ __bf16 Plds[16 * 32];   // probs tile, row-major [16 q][32 k]

  const int lane = threadIdx.x;
  const int n16  = lane & 15;
  const int kh   = lane >> 4;
  const int qi   = blockIdx.x;       // 0..127
  const int b    = blockIdx.y;       // 0..7

  const __bf16* Qb = Q + (size_t)b * NSEQ * HEAD;
  const __bf16* Kb = K + (size_t)b * NSEQ * HEAD;
  const size_t  vbase = (size_t)b * NSEQ;
  const unsigned char* padb = pad + (size_t)b * NSEQ;

  // Q A-fragments for the two K=32 steps over the 64 features
  const __bf16* qrow = Qb + (size_t)(qi * 16 + n16) * HEAD;
  bf16x16 aq0 = cat8(*(const bf16x8*)(qrow + kh * 8),
                     *(const bf16x8*)(qrow + 16 + kh * 8));
  bf16x16 aq1 = cat8(*(const bf16x8*)(qrow + 32 + kh * 8),
                     *(const bf16x8*)(qrow + 48 + kh * 8));

  const bool causal = (maskflag[0] != 0);
  const int  jmax   = causal ? ((qi * 16 + 15) >> 5) : (NSEQ / 32 - 1);

  float mrow[8], lrow[8];
  #pragma unroll
  for (int r = 0; r < 8; ++r) { mrow[r] = -__builtin_inff(); lrow[r] = 0.0f; }
  v8f accO[4] = { {}, {}, {}, {} };

  for (int j = 0; j <= jmax; ++j) {
    // ---- scores S = Qs @ K^T for 32 keys (two 16-wide column tiles)
    v8f sc[2] = { {}, {} };
    #pragma unroll
    for (int t = 0; t < 2; ++t) {
      const __bf16* krow = Kb + (size_t)(j * 32 + t * 16 + n16) * HEAD;
      bf16x16 bk0 = cat8(*(const bf16x8*)(krow + kh * 8),
                         *(const bf16x8*)(krow + 16 + kh * 8));       // feat 0..31
      bf16x16 bk1 = cat8(*(const bf16x8*)(krow + 32 + kh * 8),
                         *(const bf16x8*)(krow + 48 + kh * 8));  // feat 32..63
      sc[t] = wmma_bf16(aq0, bk0, sc[t]);
      sc[t] = wmma_bf16(aq1, bk1, sc[t]);
    }

    // ---- masks + online softmax (row stats across the 16 lanes of a half)
    const int  k0g  = j * 32 + n16;
    const int  k1g  = k0g + 16;
    const bool pad0 = padb[k0g] != 0;
    const bool pad1 = padb[k1g] != 0;

    float p0v[8], p1v[8], alph[8];
    #pragma unroll
    for (int r = 0; r < 8; ++r) {
      const int q = qi * 16 + r + 8 * kh;
      float a0 = sc[0][r], a1 = sc[1][r];
      if (pad0 || (causal && k0g > q)) a0 = -__builtin_inff();
      if (pad1 || (causal && k1g > q)) a1 = -__builtin_inff();

      float bm = fmaxf(a0, a1);
      bm = fmaxf(bm, __shfl_xor(bm, 1));
      bm = fmaxf(bm, __shfl_xor(bm, 2));
      bm = fmaxf(bm, __shfl_xor(bm, 4));
      bm = fmaxf(bm, __shfl_xor(bm, 8));

      const float mnew  = fmaxf(mrow[r], bm);
      const float alpha = __expf(mrow[r] - mnew);
      mrow[r] = mnew;

      float p0 = __expf(a0 - mnew);
      float p1 = __expf(a1 - mnew);
      float rs = p0 + p1;
      rs += __shfl_xor(rs, 1);
      rs += __shfl_xor(rs, 2);
      rs += __shfl_xor(rs, 4);
      rs += __shfl_xor(rs, 8);
      lrow[r] = lrow[r] * alpha + rs;

      alph[r] = alpha; p0v[r] = p0; p1v[r] = p1;
    }

    // ---- rescale running O accumulators
    #pragma unroll
    for (int nt = 0; nt < 4; ++nt)
      #pragma unroll
      for (int r = 0; r < 8; ++r)
        accO[nt][r] *= alph[r];

    // ---- C/D layout -> A layout via LDS
    #pragma unroll
    for (int r = 0; r < 8; ++r) {
      const int m = r + 8 * kh;
      Plds[m * 32 + n16]      = (__bf16)p0v[r];
      Plds[m * 32 + 16 + n16] = (__bf16)p1v[r];
    }
    __syncthreads();
    const __bf16* prow = Plds + n16 * 32;
    bf16x16 ap = cat8(*(const bf16x8*)(prow + kh * 8),
                      *(const bf16x8*)(prow + 16 + kh * 8));
    __syncthreads();

    // ---- O += P @ V   (V transposed: contiguous over keys)
    #pragma unroll
    for (int nt = 0; nt < 4; ++nt) {
      const __bf16* vcol = Vt + (size_t)(nt * 16 + n16) * ROWS + vbase + j * 32;
      bf16x16 bv = cat8(*(const bf16x8*)(vcol + kh * 8),
                        *(const bf16x8*)(vcol + 16 + kh * 8));
      accO[nt] = wmma_bf16(ap, bv, accO[nt]);
    }
  }

  // ---- normalize and write out (f32)
  float* ob = out + (size_t)b * NSEQ * HEAD;
  #pragma unroll
  for (int r = 0; r < 8; ++r) {
    const float inv = 1.0f / lrow[r];
    const int   row = qi * 16 + r + 8 * kh;
    #pragma unroll
    for (int nt = 0; nt < 4; ++nt)
      ob[(size_t)row * HEAD + nt * 16 + n16] = accO[nt][r] * inv;
  }
}

// ---------------------------------------------------------------------------
extern "C" void kernel_launch(void* const* d_in, const int* in_sizes, int n_in,
                              void* d_out, int out_size, void* d_ws, size_t ws_size,
                              hipStream_t stream) {
  (void)in_sizes; (void)n_in; (void)out_size; (void)ws_size;
  const float*         key_in   = (const float*)d_in[0];
  const float*         query_in = (const float*)d_in[1];
  const float*         value_in = (const float*)d_in[2];
  const unsigned char* pad      = (const unsigned char*)d_in[3];
  const int*           maskflag = (const int*)d_in[4];
  const float*         W_key    = (const float*)d_in[5];
  // d_in[6] (W_query) and d_in[7] (W_value) are unused by the reference.

  __bf16* Qws = (__bf16*)d_ws;                       // [ROWS,64] bf16, pre-scaled
  __bf16* Kws = Qws + (size_t)ROWS * HEAD;           // [ROWS,64] bf16
  __bf16* Vt  = Kws + (size_t)ROWS * HEAD;           // [64,ROWS] bf16 (transposed)
  __bf16* Wt  = Vt + (size_t)HEAD * ROWS;            // [64,512] bf16 (transposed W)

  wtrans_kernel<<<dim3((D_EMB * HEAD) / 256, 1, 1), 256, 0, stream>>>(W_key, Wt);
  proj_kernel<<<dim3(ROWS / 16, 3, 1), 32, 0, stream>>>(
      query_in, key_in, value_in, Wt, Qws, Kws, Vt);
  attn_kernel<<<dim3(NSEQ / 16, NB, 1), 32, 0, stream>>>(
      Qws, Kws, Vt, pad, maskflag, (float*)d_out);
}
